// GPT_79156247266039
// MI455X (gfx1250) — matrix operand order repaired and checked by reading, not compiled, not run on hardware
//
#include <hip/hip_runtime.h>
#include <hip/hip_bf16.h>

typedef __attribute__((ext_vector_type(16))) _Float16 v16h;
typedef __attribute__((ext_vector_type(8)))  _Float16 v8h;
typedef __attribute__((ext_vector_type(8)))  float    v8f;

#define B_   8
#define T_   1024
#define D_   256
#define NH_  8
#define HD_  32
#define E_   256
#define A_   8
#define MTOK (B_*T_)   // 8192 tokens

// ---------------------------------------------------------------- fp32 -> fp16
__global__ void k_f32tof16(const float* __restrict__ s, _Float16* __restrict__ d, int n) {
    int i = blockIdx.x * blockDim.x + threadIdx.x;
    if (i < n) d[i] = (_Float16)s[i];
}

// ---------------------------------------------------------------- embedding
// x[b,t,0:128]   = W_embed @ lcd[b,t-1]  (+pos)
// x[b,t,128:256] = W_act   @ acts[b,t]   (+pos)
__global__ void k_embed(const float* __restrict__ lcd, const float* __restrict__ acts,
                        const float* __restrict__ pos, const float* __restrict__ Wemb,
                        const float* __restrict__ Wact, float* __restrict__ x) {
    int tok = blockIdx.x;                 // 0..8191
    int b = tok / T_, t = tok % T_;
    int d = threadIdx.x;                  // 0..127
    __shared__ float sl[E_];
    __shared__ float sa[A_];
    if (t == 0) {
        sl[d] = 0.f; sl[d + 128] = 0.f;
    } else {
        const float* prev = lcd + ((size_t)b * T_ + (t - 1)) * E_;
        sl[d] = prev[d]; sl[d + 128] = prev[d + 128];
    }
    if (d < A_) sa[d] = acts[((size_t)b * T_ + t) * A_ + d];
    __syncthreads();
    float s0 = 0.f, s1 = 0.f;
    const float* we = Wemb + (size_t)d * E_;
    #pragma unroll 4
    for (int e = 0; e < E_; ++e) s0 += we[e] * sl[e];
    const float* wa = Wact + (size_t)d * A_;
    #pragma unroll
    for (int a = 0; a < A_; ++a) s1 += wa[a] * sa[a];
    float* xr = x + (size_t)tok * D_;
    xr[d]       = s0 + pos[(size_t)t * D_ + d];
    xr[d + 128] = s1 + pos[(size_t)t * D_ + d + 128];
}

// ---------------------------------------------------------------- layernorm (fp32 in, fp16 out)
__global__ void k_ln(const float* __restrict__ x, const float* __restrict__ w,
                     const float* __restrict__ bias, _Float16* __restrict__ out) {
    int tok = blockIdx.x;
    int d = threadIdx.x;                  // 0..255
    float v = x[(size_t)tok * D_ + d];
    __shared__ float r1[256], r2[256];
    r1[d] = v; r2[d] = v * v;
    __syncthreads();
    for (int s = 128; s > 0; s >>= 1) {
        if (d < s) { r1[d] += r1[d + s]; r2[d] += r2[d + s]; }
        __syncthreads();
    }
    float m   = r1[0] * (1.f / D_);
    float var = r2[0] * (1.f / D_) - m * m;
    float inv = rsqrtf(var + 1e-5f);
    out[(size_t)tok * D_ + d] = (_Float16)((v - m) * inv * w[d] + bias[d]);
}

// ---------------------------------------------------------------- WMMA GEMM
// C[M,N] = A[M,K](fp16) @ W[N,K]^T(fp16)  + bias, with fused epilogue:
//   mode 0: out fp32 = acc + bias
//   mode 1: out fp32 = acc + bias + residual
//   mode 2: out fp16 = gelu(acc + bias)
//   mode 3: out fp16 = acc + bias
// block: 8 waves -> 256 rows x 64 cols; wave = 32 rows x 64 cols
// (2 A fragments x 4 B fragments -> 8 wmma per 32-K step; each B fragment
//  feeds two WMMAs, halving B-load issue per FLOP vs a 16-row tile)
__global__ __launch_bounds__(256) void k_gemm(
        const _Float16* __restrict__ Amat, const _Float16* __restrict__ Wmat,
        const float* __restrict__ bias, const float* __restrict__ residual,
        float* __restrict__ outF, _Float16* __restrict__ outH,
        int M, int N, int K, int mode) {
    const int lane = threadIdx.x & 31;
    const int wave = threadIdx.x >> 5;
    const int m0 = blockIdx.y * 256 + wave * 32;
    const int n0 = blockIdx.x * 64;
    const int hi   = (lane & 16) ? 1 : 0;
    const int lrow = lane & 15;

    v8f acc[2][4];
    #pragma unroll
    for (int g = 0; g < 2; ++g)
        #pragma unroll
        for (int t = 0; t < 4; ++t) acc[g][t] = (v8f){};

    const _Float16* arow0 = Amat + (size_t)(m0 + lrow) * K;
    const _Float16* arow1 = Amat + (size_t)(m0 + 16 + lrow) * K;

    for (int kk = 0; kk < K; kk += 32) {
        // A fragments 16x32: lane holds row (lane&15); halves j<8 -> K=kb+j, j>=8 -> K=kb+16+(j-8)
        v16h af[2];
        {
            const _Float16* pa = arow0 + kk + (hi ? 8 : 0);
            v8h lo = *(const v8h*)pa;
            v8h hh = *(const v8h*)(pa + 16);
            #pragma unroll
            for (int j = 0; j < 8; ++j) { af[0][j] = lo[j]; af[0][j + 8] = hh[j]; }
        }
        {
            const _Float16* pa = arow1 + kk + (hi ? 8 : 0);
            v8h lo = *(const v8h*)pa;
            v8h hh = *(const v8h*)(pa + 16);
            #pragma unroll
            for (int j = 0; j < 8; ++j) { af[1][j] = lo[j]; af[1][j + 8] = hh[j]; }
        }
        // B fragments 32x16: lane holds col (lane&15); same K order as A (8 at +0, 8 at +16)
        #pragma unroll
        for (int t = 0; t < 4; ++t) {
            const _Float16* pw = Wmat + (size_t)(n0 + t * 16 + lrow) * K + kk + (hi ? 8 : 0);
            v16h bf;
            {
                v8h lo = *(const v8h*)pw;
                v8h hh = *(const v8h*)(pw + 16);
                #pragma unroll
                for (int j = 0; j < 8; ++j) { bf[j] = lo[j]; bf[j + 8] = hh[j]; }
            }
            acc[0][t] = __builtin_amdgcn_wmma_f32_16x16x32_f16(false, af[0], false, bf,
                                                               (short)0, acc[0][t], false, false);
            acc[1][t] = __builtin_amdgcn_wmma_f32_16x16x32_f16(false, af[1], false, bf,
                                                               (short)0, acc[1][t], false, false);
        }
    }

    #pragma unroll
    for (int g = 0; g < 2; ++g) {
        #pragma unroll
        for (int t = 0; t < 4; ++t) {
            int col = n0 + t * 16 + lrow;
            float bv = bias[col];
            #pragma unroll
            for (int r = 0; r < 8; ++r) {
                int row = m0 + g * 16 + r + (hi ? 8 : 0);
                size_t idx = (size_t)row * N + col;
                float v = acc[g][t][r] + bv;
                if (mode == 1) v += residual[idx];
                if (mode == 2) v = 0.5f * v * (1.f + erff(v * 0.70710678118654752f));
                if (mode <= 1) outF[idx] = v;
                else           outH[idx] = (_Float16)v;
            }
        }
    }
}

// ---------------------------------------------------------------- V transpose: v[b,t,h,d] -> vt[b,h,d,t]
__global__ void k_vtr(const _Float16* __restrict__ v, _Float16* __restrict__ vt) {
    int i = blockIdx.x * blockDim.x + threadIdx.x;     // over MTOK*D_
    int tok = i / D_, dd = i % D_;
    int b = tok / T_, t = tok % T_;
    int h = dd / HD_, d = dd % HD_;
    vt[((size_t)(b * NH_ + h) * HD_ + d) * T_ + t] = v[i];
}

// ---------------------------------------------------------------- fused causal flash attention
// grid (T/64, B*NH), block 128 = 4 waves; each wave owns one 16-query tile.
__global__ __launch_bounds__(128) void k_attn(
        const _Float16* __restrict__ Q, const _Float16* __restrict__ Km,
        const _Float16* __restrict__ Vt, _Float16* __restrict__ Y) {
    const int lane = threadIdx.x & 31;
    const int wave = threadIdx.x >> 5;
    const int bh = blockIdx.y;
    const int b = bh / NH_, h = bh % NH_;
    const int q0 = (blockIdx.x * 4 + wave) * 16;
    const int lrow = lane & 15;
    const int hi = (lane & 16) ? 1 : 0;

    __shared__ __align__(16) _Float16 pls[4][16 * 32];
    _Float16* myp = pls[wave];

    const _Float16* qbase = Q  + ((size_t)b * T_) * D_ + h * HD_;
    const _Float16* kbase = Km + ((size_t)b * T_) * D_ + h * HD_;
    const _Float16* vtb   = Vt + ((size_t)(b * NH_ + h) * HD_) * T_;

    // Q fragment (16x32, K = HD)
    v16h qf;
    {
        const _Float16* pq = qbase + (size_t)(q0 + lrow) * D_ + (hi ? 8 : 0);
        v8h lo = *(const v8h*)pq;
        v8h hh = *(const v8h*)(pq + 16);
        #pragma unroll
        for (int j = 0; j < 8; ++j) { qf[j] = lo[j]; qf[j + 8] = hh[j]; }
    }

    v8f o0 = {}, o1 = {};
    float mi[8], li[8];
    #pragma unroll
    for (int r = 0; r < 8; ++r) { mi[r] = -3.0e38f; li[r] = 0.f; }
    const float scale = 0.17677669529663687f;   // 1/sqrt(32)

    const int nchunk = (q0 + 16 + 31) / 32;
    for (int c = 0; c < nchunk; ++c) {
        const int kb = c * 32;
        // K^T B-fragments: col = key, head dims in same order as Q (8 at +0, 8 at +16)
        const _Float16* pk0 = kbase + (size_t)(kb + lrow) * D_ + (hi ? 8 : 0);
        v16h kf0, kf1;
        {
            v8h a0 = *(const v8h*)pk0, a1 = *(const v8h*)(pk0 + 16);
            v8h b0 = *(const v8h*)(pk0 + 16 * D_), b1 = *(const v8h*)(pk0 + 16 * D_ + 16);
            #pragma unroll
            for (int j = 0; j < 8; ++j) { kf0[j] = a0[j]; kf0[j + 8] = a1[j]; kf1[j] = b0[j]; kf1[j + 8] = b1[j]; }
        }
        v8f z = {};
        v8f s0 = __builtin_amdgcn_wmma_f32_16x16x32_f16(false, qf, false, kf0, (short)0, z, false, false);
        v8f s1 = __builtin_amdgcn_wmma_f32_16x16x32_f16(false, qf, false, kf1, (short)0, z, false, false);

        float p0[8], p1[8];
        #pragma unroll
        for (int r = 0; r < 8; ++r) {
            int qrow = q0 + r + (hi ? 8 : 0);
            int k0i = kb + lrow;
            float a  = (k0i      <= qrow) ? s0[r] * scale : -3.0e38f;
            float bb = (k0i + 16 <= qrow) ? s1[r] * scale : -3.0e38f;
            float rmax = fmaxf(a, bb);
            #pragma unroll
            for (int m = 1; m < 16; m <<= 1) rmax = fmaxf(rmax, __shfl_xor(rmax, m, 32));
            float mnew = fmaxf(mi[r], rmax);
            float e0 = expf(a - mnew), e1 = expf(bb - mnew);
            float rs = e0 + e1;
            #pragma unroll
            for (int m = 1; m < 16; m <<= 1) rs += __shfl_xor(rs, m, 32);
            float corr = expf(mi[r] - mnew);
            li[r] = li[r] * corr + rs;
            o0[r] *= corr; o1[r] *= corr;
            mi[r] = mnew;
            p0[r] = e0; p1[r] = e1;
        }

        // restage P (C-layout) -> row-major 16x32 fp16 in per-wave LDS
        #pragma unroll
        for (int r = 0; r < 8; ++r) {
            int row = r + (hi ? 8 : 0);
            myp[row * 32 + lrow]      = (_Float16)p0[r];
            myp[row * 32 + 16 + lrow] = (_Float16)p1[r];
        }
        asm volatile("" ::: "memory");
        __builtin_amdgcn_wave_barrier();   // wave-local LDS pipe is in-order; block reordering

        // P as A-fragment (16 queries x 32 keys)
        v16h pf;
        {
            const _Float16* pp = myp + lrow * 32 + (hi ? 8 : 0);
            v8h lo = *(const v8h*)pp;
            v8h hh = *(const v8h*)(pp + 16);
            #pragma unroll
            for (int j = 0; j < 8; ++j) { pf[j] = lo[j]; pf[j + 8] = hh[j]; }
        }
        asm volatile("" ::: "memory");
        __builtin_amdgcn_wave_barrier();

        // V B-fragments from V^T (keys in same order as P: 8 at +0, 8 at +16)
        const _Float16* pv0 = vtb + (size_t)lrow * T_ + kb + (hi ? 8 : 0);
        v16h vf0, vf1;
        {
            v8h a0 = *(const v8h*)pv0, a1 = *(const v8h*)(pv0 + 16);
            v8h b0 = *(const v8h*)(pv0 + 16 * T_), b1 = *(const v8h*)(pv0 + 16 * T_ + 16);
            #pragma unroll
            for (int j = 0; j < 8; ++j) { vf0[j] = a0[j]; vf0[j + 8] = a1[j]; vf1[j] = b0[j]; vf1[j + 8] = b1[j]; }
        }
        o0 = __builtin_amdgcn_wmma_f32_16x16x32_f16(false, pf, false, vf0, (short)0, o0, false, false);
        o1 = __builtin_amdgcn_wmma_f32_16x16x32_f16(false, pf, false, vf1, (short)0, o1, false, false);
    }

    _Float16* yrow = Y + ((size_t)b * T_) * D_ + h * HD_;
    #pragma unroll
    for (int r = 0; r < 8; ++r) {
        int qrow = q0 + r + (hi ? 8 : 0);
        float invl = 1.f / li[r];
        yrow[(size_t)qrow * D_ + lrow]      = (_Float16)(o0[r] * invl);
        yrow[(size_t)qrow * D_ + 16 + lrow] = (_Float16)(o1[r] * invl);
    }
}

// ---------------------------------------------------------------- host orchestration
extern "C" void kernel_launch(void* const* d_in, const int* in_sizes, int n_in,
                              void* d_out, int out_size, void* d_ws, size_t ws_size,
                              hipStream_t stream) {
    (void)in_sizes; (void)n_in; (void)out_size; (void)ws_size;
    const float* lcd  = (const float*)d_in[0];
    const float* acts = (const float*)d_in[1];
    const float* pos  = (const float*)d_in[2];
    const float* Wemb = (const float*)d_in[3];
    const float* Wact = (const float*)d_in[4];
    const float* ln1w = (const float*)d_in[5];
    const float* ln1b = (const float*)d_in[6];
    const float* Wq = (const float*)d_in[7];   const float* bq = (const float*)d_in[8];
    const float* Wk = (const float*)d_in[9];   const float* bk = (const float*)d_in[10];
    const float* Wv = (const float*)d_in[11];  const float* bv = (const float*)d_in[12];
    const float* Wp = (const float*)d_in[13];  const float* bp = (const float*)d_in[14];
    const float* ln2w = (const float*)d_in[15]; const float* ln2b = (const float*)d_in[16];
    const float* W1 = (const float*)d_in[17];  const float* b1 = (const float*)d_in[18];
    const float* W2 = (const float*)d_in[19];  const float* b2 = (const float*)d_in[20];
    const float* lnfw = (const float*)d_in[21]; const float* lnfb = (const float*)d_in[22];
    const float* Wh = (const float*)d_in[23];  const float* bh = (const float*)d_in[24];

    char* ws = (char*)d_ws;
    float*    x    = (float*)   (ws + 0);            //  8 MiB  fp32 residual stream
    _Float16* hbuf = (_Float16*)(ws + 8388608);      //  4 MiB  LN output
    _Float16* qbuf = (_Float16*)(ws + 12582912);     //  4 MiB
    _Float16* kbuf = (_Float16*)(ws + 16777216);     //  4 MiB
    _Float16* vbuf = (_Float16*)(ws + 20971520);     //  4 MiB
    _Float16* vtbf = (_Float16*)(ws + 25165824);     //  4 MiB  V^T
    _Float16* ybuf = (_Float16*)(ws + 29360128);     //  4 MiB  attention out
    _Float16* h2   = (_Float16*)(ws + 33554432);     // 16 MiB  MLP mid
    _Float16* Wq16 = (_Float16*)(ws + 50331648);
    _Float16* Wk16 = (_Float16*)(ws + 50855936);
    _Float16* Wv16 = (_Float16*)(ws + 51380224);
    _Float16* Wp16 = (_Float16*)(ws + 51904512);
    _Float16* W116 = (_Float16*)(ws + 52428800);
    _Float16* W216 = (_Float16*)(ws + 54525952);
    _Float16* Wh16 = (_Float16*)(ws + 56623104);     // end ~56.8 MB

    auto conv = [&](const float* s, _Float16* d, int n) {
        k_f32tof16<<<(n + 255) / 256, 256, 0, stream>>>(s, d, n);
    };
    conv(Wq, Wq16, 4 * D_ * D_);
    conv(Wk, Wk16, 4 * D_ * D_);
    conv(Wv, Wv16, 4 * D_ * D_);
    conv(Wp, Wp16, 4 * D_ * D_);
    conv(W1, W116, 4 * 4 * D_ * D_);
    conv(W2, W216, 4 * 4 * D_ * D_);
    conv(Wh, Wh16, E_ * D_);

    auto gemm = [&](const _Float16* Am, const _Float16* Wm, const float* bias,
                    const float* res, float* oF, _Float16* oH, int N, int K, int mode) {
        dim3 g(N / 64, MTOK / 256);
        k_gemm<<<g, 256, 0, stream>>>(Am, Wm, bias, res, oF, oH, MTOK, N, K, mode);
    };

    k_embed<<<MTOK, 128, 0, stream>>>(lcd, acts, pos, Wemb, Wact, x);

    for (int i = 0; i < 4; ++i) {
        k_ln<<<MTOK, 256, 0, stream>>>(x, ln1w + i * D_, ln1b + i * D_, hbuf);
        gemm(hbuf, Wq16 + i * D_ * D_, bq + i * D_, nullptr, nullptr, qbuf, D_, D_, 3);
        gemm(hbuf, Wk16 + i * D_ * D_, bk + i * D_, nullptr, nullptr, kbuf, D_, D_, 3);
        gemm(hbuf, Wv16 + i * D_ * D_, bv + i * D_, nullptr, nullptr, vbuf, D_, D_, 3);
        k_vtr<<<(MTOK * D_) / 256, 256, 0, stream>>>(vbuf, vtbf);
        k_attn<<<dim3(T_ / 64, B_ * NH_), 128, 0, stream>>>(qbuf, kbuf, vtbf, ybuf);
        gemm(ybuf, Wp16 + i * D_ * D_, bp + i * D_, x, x, nullptr, D_, D_, 1);
        k_ln<<<MTOK, 256, 0, stream>>>(x, ln2w + i * D_, ln2b + i * D_, hbuf);
        gemm(hbuf, W116 + i * 4 * D_ * D_, b1 + i * 4 * D_, nullptr, nullptr, h2, 4 * D_, D_, 2);
        gemm(h2, W216 + i * 4 * D_ * D_, b2 + i * D_, x, x, nullptr, D_, 4 * D_, 1);
    }

    k_ln<<<MTOK, 256, 0, stream>>>(x, lnfw, lnfb, hbuf);
    gemm(hbuf, Wh16, bh, nullptr, (float*)d_out, nullptr, E_, D_, 0);
}
